// MultiHeadedAttention_18270790877398
// MI455X (gfx1250) — matrix operand order repaired and checked by reading, not compiled, not run on hardware
//
#include <hip/hip_runtime.h>
#include <hip/hip_bf16.h>

typedef __attribute__((ext_vector_type(16))) __bf16 v16bf;
typedef __attribute__((ext_vector_type(8)))  __bf16 v8bf;
typedef __attribute__((ext_vector_type(8)))  float  v8f;

#define DMODEL 1024
#define SEQ    2048
#define NB     4
#define NH     16
#define DK     64
#define MROWS  (NB * SEQ)   // 8192

__device__ __forceinline__ __bf16 f2bf(float f) {
  unsigned u = __builtin_bit_cast(unsigned, f);
  unsigned r = u + 0x7FFFu + ((u >> 16) & 1u);     // round-to-nearest-even
  unsigned short h = (unsigned short)(r >> 16);
  return __builtin_bit_cast(__bf16, h);
}

__device__ __forceinline__ v8f wmma_bf16(v16bf a, v16bf b, v8f c) {
  return __builtin_amdgcn_wmma_f32_16x16x32_bf16(false, a, false, b, (short)0, c,
                                                 false, false);
}

__device__ __forceinline__ v16bf cat16(v8bf lo, v8bf hi) {
  return __builtin_shufflevector(lo, hi, 0, 1, 2, 3, 4, 5, 6, 7,
                                 8, 9, 10, 11, 12, 13, 14, 15);
}

__global__ void cvt_bf16(const float* __restrict__ in, __bf16* __restrict__ out, int n) {
  int i = blockIdx.x * blockDim.x + threadIdx.x;
  if (i < n) out[i] = f2bf(in[i]);
}

// Pre-pack W [1024,1024] f32 into per-lane WMMA B tiles, bf16, contiguous k order:
// out[((kb*64 + nt)*32 + lane)*16 + e] = W[kb*32 + 16*(lane>>4) + e][nt*16 + (lane&15)]
__global__ void pack_w(const float* __restrict__ W, __bf16* __restrict__ out) {
  int i = blockIdx.x * blockDim.x + threadIdx.x;   // coalesced read
  int K = i >> 10, N = i & 1023;
  int kb = K >> 5, kk = K & 31;
  int g = kk >> 4, e = kk & 15;
  int nt = N >> 4, r = N & 15;
  out[((((size_t)kb * 64 + nt) * 32) + g * 16 + r) * 16 + e] = f2bf(W[i]);
}

// C[M=8192, N=1024] = A(bf16 row-major) @ W(pre-packed B-layout) + bias
// MODE 0: Q -> [B,H,S,Dk] * 0.125    MODE 1: K -> [B,H,S,Dk]
// MODE 2: V -> [B,H,Dk,S] transposed MODE 3: fp32 out [M,N]
template <int MODE>
__global__ __launch_bounds__(256) void gemm16(const __bf16* __restrict__ A,
                                              const __bf16* __restrict__ PW,
                                              const float* __restrict__ bias,
                                              void* __restrict__ outp) {
  const int lane = threadIdx.x & 31;
  const int wave = threadIdx.x >> 5;
  const int g = lane >> 4, r = lane & 15;
  const int mBase = blockIdx.y * 16;
  const int nt = blockIdx.x * 8 + wave;
  const int nBase = nt * 16;
  const __bf16* arow = A + (size_t)(mBase + r) * DMODEL + 16 * g;
  const __bf16* wp = PW + ((size_t)nt * 32 + lane) * 16;
  v8f acc = {};
#pragma unroll 4
  for (int k0 = 0; k0 < DMODEL; k0 += 32) {
    v8bf alo = *(const v8bf*)(arow + k0);
    v8bf ahi = *(const v8bf*)(arow + k0 + 8);
    v16bf a = cat16(alo, ahi);
    v16bf b = *(const v16bf*)(wp + (size_t)(k0 >> 5) * (64 * 32 * 16));
    acc = wmma_bf16(a, b, acc);
  }
  const float bval = bias[nBase + r];
  if (MODE == 3) {
#pragma unroll
    for (int i = 0; i < 8; ++i) {
      const int m = mBase + i + 8 * g;
      ((float*)outp)[(size_t)m * DMODEL + nBase + r] = acc[i] + bval;
    }
  } else if (MODE == 2) {   // V transposed [B,H,Dk,S]: 8 consecutive s per lane
    const int n = nBase + r;
    const int h = n >> 6, d = n & (DK - 1);
    const int b = mBase >> 11;
    const int sBase = (mBase & (SEQ - 1)) + 8 * g;
    v8bf pk;
#pragma unroll
    for (int i = 0; i < 8; ++i) pk[i] = f2bf(acc[i] + bval);
    *(v8bf*)((__bf16*)outp + ((size_t)(b * NH + h) * DK + d) * SEQ + sBase) = pk;
  } else {                  // Q (scaled) / K row-major [B,H,S,Dk]
    const int n = nBase + r;
    const int h = n >> 6, d = n & (DK - 1);
#pragma unroll
    for (int i = 0; i < 8; ++i) {
      const int m = mBase + i + 8 * g;
      const int b = m >> 11, s = m & (SEQ - 1);
      float v = acc[i] + bval;
      if (MODE == 0) v *= 0.125f;
      ((__bf16*)outp)[((size_t)(b * NH + h) * SEQ + s) * DK + d] = f2bf(v);
    }
  }
}

// Flash attention via S^T = K*Q^T, O^T = V^T*P^T. One wave per (b,h,16-query tile),
// 32-key blocks, causal. Fixed query per lane -> scalar m/l, 1 shfl per reduction.
// Q,K row-major [B,H,S,Dk]; V transposed [B,H,Dk,S].
__global__ __launch_bounds__(256) void attn16(const __bf16* __restrict__ Qh,
                                              const __bf16* __restrict__ Kh,
                                              const __bf16* __restrict__ VT,
                                              __bf16* __restrict__ Out) {
  __shared__ __bf16 plds[8][16 * 32];   // per-wave P^T patch: [query r][key 0..31]
  const int lane = threadIdx.x & 31;
  const int wave = threadIdx.x >> 5;
  const int g = lane >> 4, r = lane & 15;
  const int tile = blockIdx.x * 8 + wave;
  const int bh = tile >> 7;
  const int qt = tile & 127;
  const int qBase = qt * 16;
  const __bf16* Qp = Qh + (size_t)bh * SEQ * DK;
  const __bf16* Kp = Kh + (size_t)bh * SEQ * DK;
  const __bf16* Vp = VT + (size_t)bh * DK * SEQ;

  // Persistent B tiles: Q^T, b[e] = Q[qBase+r][c*32 + 16g + e] (contiguous)
  v16bf bq[2];
#pragma unroll
  for (int c = 0; c < 2; ++c)
    bq[c] = *(const v16bf*)(Qp + (size_t)(qBase + r) * DK + c * 32 + 16 * g);

  v8f o[4] = {};               // O^T tiles: M = d (t*16+i+8g), N = query r
  float m = -3.0e38f, l = 0.0f;
  const int q = qBase + r;     // this lane's query index

  const int nkb = (qBase + 15) / 32 + 1;
  for (int kb = 0; kb < nkb; ++kb) {
    const int keyBase = kb * 32;
    v8f s0 = {}, s1 = {};      // S^T: s0 keys keyBase+i+8g, s1 keys keyBase+16+i+8g
#pragma unroll
    for (int c = 0; c < 2; ++c) {
      // A = K tiles (rows = keys), piecewise-contiguous from row-major K
      v8bf k0lo = *(const v8bf*)(Kp + (size_t)(keyBase + r) * DK + c * 32 + 16 * g);
      v8bf k0hi = *(const v8bf*)(Kp + (size_t)(keyBase + r) * DK + c * 32 + 8 + 16 * g);
      v8bf k1lo = *(const v8bf*)(Kp + (size_t)(keyBase + 16 + r) * DK + c * 32 + 16 * g);
      v8bf k1hi = *(const v8bf*)(Kp + (size_t)(keyBase + 16 + r) * DK + c * 32 + 8 + 16 * g);
      s0 = wmma_bf16(cat16(k0lo, k0hi), bq[c], s0);
      s1 = wmma_bf16(cat16(k1lo, k1hi), bq[c], s1);
    }
    if (keyBase + 31 > qBase) {   // causal mask: key > query -> -inf
#pragma unroll
      for (int i = 0; i < 8; ++i) {
        if (keyBase + i + 8 * g > q)      s0[i] = -3.0e38f;
        if (keyBase + 16 + i + 8 * g > q) s1[i] = -3.0e38f;
      }
    }
    // row max over this lane's 16 keys, then combine with the other half
    float mi = fmaxf(s0[0], s1[0]);
#pragma unroll
    for (int i = 1; i < 8; ++i) mi = fmaxf(mi, fmaxf(s0[i], s1[i]));
    mi = fmaxf(mi, __shfl_xor(mi, 16, 32));
    const float mnew = fmaxf(m, mi);
    v8bf pk0, pk1;
    float ls = 0.0f;
#pragma unroll
    for (int i = 0; i < 8; ++i) {
      const float e0 = __expf(s0[i] - mnew);
      const float e1 = __expf(s1[i] - mnew);
      ls += e0 + e1;
      pk0[i] = f2bf(e0);
      pk1[i] = f2bf(e1);
    }
    ls += __shfl_xor(ls, 16, 32);
    const float sc = __expf(m - mnew);
    l = l * sc + ls;
    m = mnew;
#pragma unroll
    for (int t = 0; t < 4; ++t)
#pragma unroll
      for (int i = 0; i < 8; ++i) o[t][i] *= sc;
    // P^T to LDS: per-lane contiguous 8-elem runs -> two b128 stores
    *(v8bf*)&plds[wave][r * 32 + 8 * g]      = pk0;  // keys 8g..8g+7
    *(v8bf*)&plds[wave][r * 32 + 16 + 8 * g] = pk1;  // keys 16+8g..23+8g
    asm volatile("s_wait_dscnt 0" ::: "memory");
    // B = P^T: b[e] = plds[r*32 + 16g + e] (contiguous 32B)
    v16bf bp = *(const v16bf*)&plds[wave][r * 32 + 16 * g];
#pragma unroll
    for (int t = 0; t < 4; ++t) {
      // A = V^T tiles (rows = d), piecewise-contiguous from [Dk,S]
      v8bf vlo = *(const v8bf*)(Vp + (size_t)(t * 16 + r) * SEQ + keyBase + 16 * g);
      v8bf vhi = *(const v8bf*)(Vp + (size_t)(t * 16 + r) * SEQ + keyBase + 8 + 16 * g);
      o[t] = wmma_bf16(cat16(vlo, vhi), bp, o[t]);
    }
  }
  // O^T epilogue: fixed sequence row per lane, contiguous d columns -> packed stores
  const float inv = 1.0f / l;
  const int b = bh >> 4, h = bh & 15;
  __bf16* outRow = Out + (size_t)(b * SEQ + qBase + r) * DMODEL + h * DK;
#pragma unroll
  for (int t = 0; t < 4; ++t) {
    v8bf pk;
#pragma unroll
    for (int i = 0; i < 8; ++i) pk[i] = f2bf(o[t][i] * inv);
    *(v8bf*)(outRow + t * 16 + 8 * g) = pk;
  }
}

extern "C" void kernel_launch(void* const* d_in, const int* in_sizes, int n_in,
                              void* d_out, int out_size, void* d_ws, size_t ws_size,
                              hipStream_t stream) {
  (void)in_sizes; (void)n_in; (void)out_size; (void)ws_size;
  const float* x  = (const float*)d_in[0];
  const float* Wq = (const float*)d_in[1];
  const float* bq = (const float*)d_in[2];
  const float* Wk = (const float*)d_in[3];
  const float* bk = (const float*)d_in[4];
  const float* Wv = (const float*)d_in[5];
  const float* bv = (const float*)d_in[6];
  const float* Wo = (const float*)d_in[7];
  const float* bo = (const float*)d_in[8];

  char* ws = (char*)d_ws;
  size_t off = 0;
  auto alloc = [&](size_t bytes) {
    void* p = ws + off;
    off += (bytes + 255) & ~(size_t)255;
    return p;
  };
  const size_t nX = (size_t)MROWS * DMODEL;   // 8,388,608
  const size_t nW = (size_t)DMODEL * DMODEL;  // 1,048,576
  __bf16* xh  = (__bf16*)alloc(nX * 2);
  __bf16* Wqh = (__bf16*)alloc(nW * 2);   // packed B-layout
  __bf16* Wkh = (__bf16*)alloc(nW * 2);
  __bf16* Wvh = (__bf16*)alloc(nW * 2);
  __bf16* Woh = (__bf16*)alloc(nW * 2);
  __bf16* Qh  = (__bf16*)alloc(nX * 2);   // [B,H,S,Dk], pre-scaled by 1/8
  __bf16* Kh  = (__bf16*)alloc(nX * 2);   // [B,H,S,Dk]
  __bf16* VT  = (__bf16*)alloc(nX * 2);   // [B,H,Dk,S]
  __bf16* AO  = (__bf16*)alloc(nX * 2);   // [B,S,D]

  cvt_bf16<<<(int)((nX + 255) / 256), 256, 0, stream>>>(x, xh, (int)nX);
  pack_w<<<(int)(nW / 256), 256, 0, stream>>>(Wq, Wqh);
  pack_w<<<(int)(nW / 256), 256, 0, stream>>>(Wk, Wkh);
  pack_w<<<(int)(nW / 256), 256, 0, stream>>>(Wv, Wvh);
  pack_w<<<(int)(nW / 256), 256, 0, stream>>>(Wo, Woh);

  dim3 gg(DMODEL / 128, MROWS / 16);   // 8 x 512 blocks, 8 waves each
  gemm16<0><<<gg, 256, 0, stream>>>(xh, Wqh, bq, Qh);
  gemm16<1><<<gg, 256, 0, stream>>>(xh, Wkh, bk, Kh);
  gemm16<2><<<gg, 256, 0, stream>>>(xh, Wvh, bv, VT);

  attn16<<<(NB * NH * (SEQ / 16)) / 8, 256, 0, stream>>>(Qh, Kh, VT, AO);

  gemm16<3><<<gg, 256, 0, stream>>>(AO, Woh, bo, d_out);
}
